// ShotMaskedMHA_27169963115067
// MI455X (gfx1250) — matrix operand order repaired and checked by reading, not compiled, not run on hardware
//
#include <hip/hip_runtime.h>
#include <hip/hip_bf16.h>

#define B_  16
#define S_  64
#define T_  4096
#define C_  512
#define H_  8
#define D_  64

typedef __attribute__((ext_vector_type(16))) __bf16 bf16x16;
typedef __attribute__((ext_vector_type(8)))  __bf16 bf16x8;
typedef __attribute__((ext_vector_type(4)))  __bf16 bf16x4;
typedef __attribute__((ext_vector_type(8)))  float  floatx8;

static __device__ inline floatx8 zero8() {
    floatx8 z;
#pragma unroll
    for (int i = 0; i < 8; ++i) z[i] = 0.f;
    return z;
}

static __device__ inline bf16x16 pack16(bf16x8 lo, bf16x8 hi) {
    bf16x16 r;
#pragma unroll
    for (int i = 0; i < 8; ++i) { r[i] = lo[i]; r[i + 8] = hi[i]; }
    return r;
}

static __device__ inline bf16x16 zbf16() {
    bf16x16 r;
#pragma unroll
    for (int i = 0; i < 16; ++i) r[i] = (__bf16)0.f;
    return r;
}

// ---------------------------------------------------------------------------
// K1: fold the single query through Wk.
//   qsh = (Wq @ x + bq) / sqrt(D)
//   wk_eff[h][c] = sum_d qsh[h*64+d] * Wk[h*64+d][c]
//   cb[h]        = sum_d qsh[h*64+d] * bk[h*64+d]
// ---------------------------------------------------------------------------
__global__ void setup_kernel(const float* __restrict__ x,
                             const float* __restrict__ Wq, const float* __restrict__ bq,
                             const float* __restrict__ Wk, const float* __restrict__ bk,
                             float* __restrict__ wk_eff, float* __restrict__ cb) {
    __shared__ float qsh[C_];
    const int tid = threadIdx.x;
    const float scale = 0.125f;  // 1/sqrt(64)
    for (int o = tid; o < C_; o += 256) {
        float acc = bq[o];
        for (int c = 0; c < C_; ++c) acc += Wq[(size_t)o * C_ + c] * x[c];
        qsh[o] = acc * scale;
    }
    __syncthreads();
    for (int idx = tid; idx < H_ * C_; idx += 256) {
        const int h = idx >> 9, c = idx & (C_ - 1);
        float acc = 0.f;
        for (int d = 0; d < D_; ++d)
            acc += qsh[h * D_ + d] * Wk[(size_t)(h * D_ + d) * C_ + c];
        wk_eff[idx] = acc;
    }
    if (tid < H_) {
        float acc = 0.f;
        for (int d = 0; d < D_; ++d) acc += qsh[tid * D_ + d] * bk[tid * D_ + d];
        cb[tid] = acc;
    }
}

// K2: convert Wv / Wp to bf16 for WMMA operands.
__global__ void convert_kernel(const float* __restrict__ Wv, const float* __restrict__ Wp,
                               __bf16* __restrict__ wvb, __bf16* __restrict__ wpb) {
    const int i = blockIdx.x * 256 + threadIdx.x;
    if (i < C_ * C_) {
        wvb[i] = (__bf16)Wv[i];
        wpb[i] = (__bf16)Wp[i];
    }
}

// ---------------------------------------------------------------------------
// K3: att[b][h][t] = wk_eff[h,:] . eh[b,:,t] + cb[h]
// Bandwidth-bound pass over eh (134 MB): each thread owns 4 consecutive t so
// every global access is b128 (512 B per wave per instruction).
// ---------------------------------------------------------------------------
__global__ void att_kernel(const float* __restrict__ eh, const float* __restrict__ wk_eff,
                           const float* __restrict__ cb, float* __restrict__ att) {
    __shared__ float wkT[C_][8];  // transposed: contiguous heads per channel
    const int tid = threadIdx.x;
    for (int i = tid; i < H_ * C_; i += 256) {
        const int h = i >> 9, c = i & (C_ - 1);
        wkT[c][h] = wk_eff[i];
    }
    __syncthreads();
    const int t0 = (blockIdx.x * 256 + tid) * 4;
    const int b = blockIdx.y;
    float acc[8][4];
#pragma unroll
    for (int h = 0; h < 8; ++h) {
        const float c0 = cb[h];
#pragma unroll
        for (int j = 0; j < 4; ++j) acc[h][j] = c0;
    }
    for (int c = 0; c < C_; ++c) {
        const float4 e = *(const float4*)&eh[((size_t)b * C_ + c) * T_ + t0];
        const float ev[4] = {e.x, e.y, e.z, e.w};
        const float4* wp4 = (const float4*)&wkT[c][0];
        const float4 w0 = wp4[0], w1 = wp4[1];
        const float wv[8] = {w0.x, w0.y, w0.z, w0.w, w1.x, w1.y, w1.z, w1.w};
#pragma unroll
        for (int h = 0; h < 8; ++h)
#pragma unroll
            for (int j = 0; j < 4; ++j) acc[h][j] += wv[h] * ev[j];
    }
#pragma unroll
    for (int h = 0; h < 8; ++h) {
        *(float4*)&att[((size_t)b * H_ + h) * T_ + t0] =
            make_float4(acc[h][0], acc[h][1], acc[h][2], acc[h][3]);
    }
}

// ---------------------------------------------------------------------------
// K4: per-(b,s) segment softmax stats (max, 1/sum) per head, wave32 reductions.
// ---------------------------------------------------------------------------
__global__ void stats_kernel(const float* __restrict__ att,
                             const int* __restrict__ sst, const int* __restrict__ sen,
                             float* __restrict__ mstat, float* __restrict__ rlstat) {
    const int s = blockIdx.x, b = blockIdx.y;
    const int st = sst[b * S_ + s], en = sen[b * S_ + s];
    const int tid = threadIdx.x, lane = tid & 31, wid = tid >> 5;
    float m[8], l[8];
#pragma unroll
    for (int h = 0; h < 8; ++h) { m[h] = -3.0e38f; l[h] = 0.f; }
    for (int t = st + tid; t < en; t += 256) {
#pragma unroll
        for (int h = 0; h < 8; ++h) {
            const float a = att[((size_t)b * H_ + h) * T_ + t];
            const float nm = fmaxf(m[h], a);
            l[h] = l[h] * __expf(m[h] - nm) + __expf(a - nm);
            m[h] = nm;
        }
    }
#pragma unroll
    for (int h = 0; h < 8; ++h) {
        for (int off = 16; off; off >>= 1) {
            const float om = __shfl_xor(m[h], off, 32);
            const float ol = __shfl_xor(l[h], off, 32);
            const float nm = fmaxf(m[h], om);
            l[h] = l[h] * __expf(m[h] - nm) + ol * __expf(om - nm);
            m[h] = nm;
        }
    }
    __shared__ float sm[8][8], sl[8][8];
    if (lane == 0) {
#pragma unroll
        for (int h = 0; h < 8; ++h) { sm[wid][h] = m[h]; sl[wid][h] = l[h]; }
    }
    __syncthreads();
    if (tid < 8) {
        const int h = tid;
        float M = -3.0e38f, L = 0.f;
        for (int w = 0; w < 8; ++w) {
            const float om = sm[w][h], ol = sl[w][h];
            const float nm = fmaxf(M, om);
            L = L * __expf(M - nm) + ol * __expf(om - nm);
            M = nm;
        }
        mstat[((size_t)b * S_ + s) * H_ + h] = M;
        rlstat[((size_t)b * S_ + s) * H_ + h] = (L > 0.f) ? (1.f / L) : 0.f;
    }
}

// ---------------------------------------------------------------------------
// K5 (WMMA): ctx[b,s,h,c] = sum_{t in segment} softmax_p[h,t] * eh[b,c,t]
// A tile (16x32 bf16): rows 0..7 = per-head weights, rows 8..15 = 0.
// B tiles (32x16 bf16): eh chunk staged in LDS (144 B rows: conflict-free
// b128 gathers since n*36 mod 64 is injective over 16 lanes).
// 64-wide t-chunks -> 8 WMMAs per wave per barrier pair.
// ---------------------------------------------------------------------------
__global__ void ctx_kernel(const float* __restrict__ eh, const float* __restrict__ att,
                           const int* __restrict__ sst, const int* __restrict__ sen,
                           const float* __restrict__ mstat, const float* __restrict__ rlstat,
                           __bf16* __restrict__ ctx) {
    __shared__ __bf16 ehs[C_][72];   // 72 KB, 144 B row stride
    __shared__ __bf16 wlds[8][64];
    __shared__ float msh[8], rlsh[8];

    const int s = blockIdx.x, b = blockIdx.y;
    const int bs = b * S_ + s;
    const int st = sst[bs], en = sen[bs];
    const int tid = threadIdx.x, lane = tid & 31, wid = tid >> 5;
    const int m16 = lane & 15, g = lane >> 4;

    if (tid < 8) {
        msh[tid] = mstat[(size_t)bs * H_ + tid];
        rlsh[tid] = rlstat[(size_t)bs * H_ + tid];
    }
    __syncthreads();

    floatx8 acc[4];
#pragma unroll
    for (int j = 0; j < 4; ++j) acc[j] = zero8();

    const int t0 = st & ~63;
    for (int tb = t0; tb < en; tb += 64) {
        // softmax weights for this chunk: two (h,k) per thread
        {
            const int h = tid >> 5;
#pragma unroll
            for (int u = 0; u < 2; ++u) {
                const int kk = (tid & 31) + 32 * u;
                const int t = tb + kk;
                float wv = 0.f;
                if (t >= st && t < en)
                    wv = __expf(att[((size_t)b * H_ + h) * T_ + t] - msh[h]) * rlsh[h];
                wlds[h][kk] = (__bf16)wv;
            }
        }
        // stage eh[b, :, tb..tb+63] -> LDS bf16; b128 global loads, b64 ds stores
        {
            const int q = tid & 15;   // float4 slot within row (t = tb + q*4)
            const int r0 = tid >> 4;  // 0..15
            for (int c = r0; c < C_; c += 16) {
                const float4 e = *(const float4*)&eh[((size_t)b * C_ + c) * T_ + tb + q * 4];
                bf16x4 v;
                v[0] = (__bf16)e.x; v[1] = (__bf16)e.y;
                v[2] = (__bf16)e.z; v[3] = (__bf16)e.w;
                *(bf16x4*)&ehs[c][q * 4] = v;
            }
        }
        if (tb + 64 < en)  // prefetch next chunk (global_prefetch_b8)
            __builtin_prefetch(&eh[((size_t)b * C_ + (tid >> 4)) * T_ + tb + 64 + (tid & 15) * 4], 0, 0);
        __syncthreads();

#pragma unroll
        for (int kc = 0; kc < 2; ++kc) {
            // A fragment: 16x32, heads on rows 0..7
            bf16x16 afrag;
            if (m16 < 8) {
                const bf16x8 alo = *(const bf16x8*)&wlds[m16][kc * 32 + 8 * g];
                const bf16x8 ahi = *(const bf16x8*)&wlds[m16][kc * 32 + 16 + 8 * g];
                afrag = pack16(alo, ahi);
            } else {
                afrag = zbf16();
            }
#pragma unroll
            for (int j = 0; j < 4; ++j) {
                const int c = (wid * 4 + j) * 16 + m16;
                const bf16x8 blo = *(const bf16x8*)&ehs[c][kc * 32 + 8 * g];
                const bf16x8 bhi = *(const bf16x8*)&ehs[c][kc * 32 + 16 + 8 * g];
                const bf16x16 bfrag = pack16(blo, bhi);
                acc[j] = __builtin_amdgcn_wmma_f32_16x16x32_bf16(
                    false, afrag, false, bfrag, (short)0, acc[j], false, false);
            }
        }
        __syncthreads();
    }
    // D layout: lane n = lane%16, VGPR r -> M = r + 8*(lane/16); heads live in M<8
    if (lane < 16) {
#pragma unroll
        for (int j = 0; j < 4; ++j) {
#pragma unroll
            for (int r = 0; r < 8; ++r) {
                const int c = (wid * 4 + j) * 16 + lane;
                ctx[((size_t)bs * H_ + r) * C_ + c] = (__bf16)acc[j][r];
            }
        }
    }
}

// ---------------------------------------------------------------------------
// K6 (WMMA): out_head[i, h*64+d] = Wv[h*64+d,:] . ctx[i,h,:] + bv
// wave = head; each wave: one 16-row M-tile x 4 N-tiles, K = 512.
// ---------------------------------------------------------------------------
__global__ void headproj_kernel(const __bf16* __restrict__ ctx, const __bf16* __restrict__ wvb,
                                const float* __restrict__ bv, __bf16* __restrict__ oh) {
    const int tid = threadIdx.x, lane = tid & 31, h = tid >> 5;
    const int m16 = lane & 15, g = lane >> 4;
    const int mrow = blockIdx.x * 16;

    floatx8 acc[4];
#pragma unroll
    for (int j = 0; j < 4; ++j) acc[j] = zero8();

    const __bf16* arow = ctx + ((size_t)(mrow + m16) * H_ + h) * C_;
    for (int k0 = 0; k0 < C_; k0 += 32) {
        const bf16x8 alo = *(const bf16x8*)(arow + k0 + 8 * g);
        const bf16x8 ahi = *(const bf16x8*)(arow + k0 + 16 + 8 * g);
        const bf16x16 afrag = pack16(alo, ahi);
#pragma unroll
        for (int j = 0; j < 4; ++j) {
            const __bf16* brow = wvb + (size_t)(h * 64 + j * 16 + m16) * C_;
            const bf16x8 blo = *(const bf16x8*)(brow + k0 + 8 * g);
            const bf16x8 bhi = *(const bf16x8*)(brow + k0 + 16 + 8 * g);
            const bf16x16 bfrag = pack16(blo, bhi);
            acc[j] = __builtin_amdgcn_wmma_f32_16x16x32_bf16(
                false, afrag, false, bfrag, (short)0, acc[j], false, false);
        }
    }
#pragma unroll
    for (int j = 0; j < 4; ++j) {
        const int col = h * 64 + j * 16 + m16;
        const float bias = bv[col];
#pragma unroll
        for (int r = 0; r < 8; ++r) {
            const int i = mrow + r + 8 * g;
            oh[(size_t)i * C_ + col] = (__bf16)(acc[j][r] + bias);
        }
    }
}

// ---------------------------------------------------------------------------
// K7 (WMMA): y = out_head @ Wp^T + bp  -> f32 output (B*S, C, 1)
// ---------------------------------------------------------------------------
__global__ void outproj_kernel(const __bf16* __restrict__ oh, const __bf16* __restrict__ wpb,
                               const float* __restrict__ bp, float* __restrict__ y) {
    const int tid = threadIdx.x, lane = tid & 31, wid = tid >> 5;
    const int m16 = lane & 15, g = lane >> 4;
    const int mrow = blockIdx.x * 16;
    const int ncol0 = wid * 64;

    floatx8 acc[4];
#pragma unroll
    for (int j = 0; j < 4; ++j) acc[j] = zero8();

    const __bf16* arow = oh + (size_t)(mrow + m16) * C_;
    for (int k0 = 0; k0 < C_; k0 += 32) {
        const bf16x8 alo = *(const bf16x8*)(arow + k0 + 8 * g);
        const bf16x8 ahi = *(const bf16x8*)(arow + k0 + 16 + 8 * g);
        const bf16x16 afrag = pack16(alo, ahi);
#pragma unroll
        for (int j = 0; j < 4; ++j) {
            const __bf16* brow = wpb + (size_t)(ncol0 + j * 16 + m16) * C_;
            const bf16x8 blo = *(const bf16x8*)(brow + k0 + 8 * g);
            const bf16x8 bhi = *(const bf16x8*)(brow + k0 + 16 + 8 * g);
            const bf16x16 bfrag = pack16(blo, bhi);
            acc[j] = __builtin_amdgcn_wmma_f32_16x16x32_bf16(
                false, afrag, false, bfrag, (short)0, acc[j], false, false);
        }
    }
#pragma unroll
    for (int j = 0; j < 4; ++j) {
        const int col = ncol0 + j * 16 + m16;
        const float bias = bp[col];
#pragma unroll
        for (int r = 0; r < 8; ++r) {
            const int i = mrow + r + 8 * g;
            y[(size_t)i * C_ + col] = acc[j][r] + bias;
        }
    }
}

extern "C" void kernel_launch(void* const* d_in, const int* in_sizes, int n_in,
                              void* d_out, int out_size, void* d_ws, size_t ws_size,
                              hipStream_t stream) {
    (void)in_sizes; (void)n_in; (void)out_size; (void)ws_size;
    const float* x   = (const float*)d_in[0];
    const float* eh  = (const float*)d_in[1];
    const int*   sst = (const int*)d_in[2];
    const int*   sen = (const int*)d_in[3];
    const float* Wq  = (const float*)d_in[4];
    const float* bq  = (const float*)d_in[5];
    const float* Wk  = (const float*)d_in[6];
    const float* bk  = (const float*)d_in[7];
    const float* Wv  = (const float*)d_in[8];
    const float* bv  = (const float*)d_in[9];
    const float* Wp  = (const float*)d_in[10];
    const float* bp  = (const float*)d_in[11];
    float* y = (float*)d_out;

    // workspace layout (≈12.1 MB, 256B-aligned sections)
    char* w = (char*)d_ws;
    float*  wk_eff = (float*)(w + 0);          // 8*512 f32      = 16384 B
    float*  cb     = (float*)(w + 16384);      // 8 f32 (padded) = 256 B
    float*  mstat  = (float*)(w + 16640);      // B*S*H f32      = 32768 B
    float*  rlstat = (float*)(w + 49408);      // B*S*H f32      = 32768 B
    float*  att    = (float*)(w + 82176);      // B*H*T f32      = 2097152 B
    __bf16* ctx    = (__bf16*)(w + 2179328);   // B*S*H*C bf16   = 8388608 B
    __bf16* oh     = (__bf16*)(w + 10567936);  // B*S*C bf16     = 1048576 B
    __bf16* wvb    = (__bf16*)(w + 11616512);  // C*C bf16       = 524288 B
    __bf16* wpb    = (__bf16*)(w + 12140800);  // C*C bf16       = 524288 B

    setup_kernel<<<1, 256, 0, stream>>>(x, Wq, bq, Wk, bk, wk_eff, cb);
    convert_kernel<<<(C_ * C_) / 256, 256, 0, stream>>>(Wv, Wp, wvb, wpb);
    att_kernel<<<dim3(T_ / 1024, B_), 256, 0, stream>>>(eh, wk_eff, cb, att);
    stats_kernel<<<dim3(S_, B_), 256, 0, stream>>>(att, sst, sen, mstat, rlstat);
    ctx_kernel<<<dim3(S_, B_), 256, 0, stream>>>(eh, att, sst, sen, mstat, rlstat, ctx);
    headproj_kernel<<<(B_ * S_) / 16, 256, 0, stream>>>(ctx, wvb, bv, oh);
    outproj_kernel<<<(B_ * S_) / 16, 256, 0, stream>>>(oh, wpb, bp, y);
}
